// DistLoss_8641474199881
// MI455X (gfx1250) — matrix operand order repaired and checked by reading, not compiled, not run on hardware
//
#include <hip/hip_runtime.h>

// ---------------------------------------------------------------------------
// DistLoss on MI455X (gfx1250, wave32):
//   loss = (sum_{i,n} d[i,n] - 2 * sum_i d[i,i]) / B * 0.1
//   d[i,n] = sqrt(max(||x_i||^2 + ||y_n||^2 - 2 x_i.y_n, 0))
// GEMM x@y^T done with v_wmma_f32_16x16x32_bf16 using split-bf16 (hi+lo,
// 3 passes) for near-fp32 accuracy; distance matrix is reduced in-register.
// ---------------------------------------------------------------------------

#define BDIM 8192
#define DDIM 128
#define BT   64     // output tile edge per workgroup
#define SA   72     // padded LDS row stride (bf16 elements); 144B keeps 16B align
#define WEIGHT 0.1

typedef __attribute__((ext_vector_type(16))) __bf16 v16bf;
typedef __attribute__((ext_vector_type(8)))  float  v8f;

union FragBF {
    uint4 u[2];
    v16bf v;
};

// fp32 -> bf16 round-to-nearest-even (bit pattern)
static __device__ __forceinline__ unsigned short f2bf(float x) {
    unsigned u = __float_as_uint(x);
    unsigned r = u + (0x7FFFu + ((u >> 16) & 1u));
    return (unsigned short)(r >> 16);
}
static __device__ __forceinline__ float bf2f(unsigned short h) {
    return __uint_as_float(((unsigned)h) << 16);
}

// ---------------------------------------------------------------------------
// Kernel 1: row squared-norms for both matrices; one wave32 per row.
// rows [0,8192) -> xx from output, rows [8192,16384) -> yy from target.
// Also zeroes the global f64 accumulator (safe: main kernel is stream-ordered
// after this whole kernel completes).
// ---------------------------------------------------------------------------
__global__ __launch_bounds__(256) void distloss_norms(
        const float* __restrict__ outp, const float* __restrict__ tgtp,
        float* __restrict__ xx, float* __restrict__ yy,
        double* __restrict__ acc)
{
    const int tid  = threadIdx.x;
    const int lane = tid & 31;
    const int row  = blockIdx.x * 8 + (tid >> 5);      // 2048 blocks * 8 waves

    if (blockIdx.x == 0 && tid == 0) *acc = 0.0;

    const float* src = (row < BDIM) ? (outp + (size_t)row * DDIM)
                                    : (tgtp + (size_t)(row - BDIM) * DDIM);
    float4 v = ((const float4*)src)[lane];              // 32 lanes * 4 = 128
    float s = v.x * v.x + v.y * v.y + v.z * v.z + v.w * v.w;
    #pragma unroll
    for (int m = 16; m >= 1; m >>= 1)
        s += __shfl_xor(s, m, 32);
    if (lane == 0) {
        if (row < BDIM) xx[row] = s;
        else            yy[row - BDIM] = s;
    }
}

// ---------------------------------------------------------------------------
// Kernel 2: 64x64 tile of the distance matrix per workgroup (256 thr / 8 waves)
// ---------------------------------------------------------------------------
__global__ __launch_bounds__(256) void distloss_tile(
        const float* __restrict__ outp, const float* __restrict__ tgtp,
        const float* __restrict__ xx,   const float* __restrict__ yy,
        double* __restrict__ acc)
{
    __shared__ unsigned short Ahi[BT * SA];
    __shared__ unsigned short Alo[BT * SA];
    __shared__ unsigned short Bhi[BT * SA];
    __shared__ unsigned short Blo[BT * SA];
    __shared__ float red[256];

    const int tid = threadIdx.x;
    const int bn  = blockIdx.x;   // column tile (target rows)
    const int bm  = blockIdx.y;   // row tile (output rows)

    // ---- cooperative load + fp32 -> (hi,lo) bf16 split into LDS ----
    const float* ag = outp + (size_t)bm * BT * DDIM;    // 64x128 contiguous
    const float* bg = tgtp + (size_t)bn * BT * DDIM;
    #pragma unroll
    for (int i = 0; i < 8; ++i) {
        int idx4 = tid + i * 256;                       // 0..2047 float4s
        int off  = idx4 * 4;                            // element in panel
        int r    = off >> 7;                            // /128
        int c    = off & 127;
        float4 av = ((const float4*)ag)[idx4];
        float4 bv = ((const float4*)bg)[idx4];

        unsigned short ah[4], al[4], bh[4], bl[4];
        float af[4] = {av.x, av.y, av.z, av.w};
        float bf[4] = {bv.x, bv.y, bv.z, bv.w};
        #pragma unroll
        for (int j = 0; j < 4; ++j) {
            ah[j] = f2bf(af[j]);
            al[j] = f2bf(af[j] - bf2f(ah[j]));
            bh[j] = f2bf(bf[j]);
            bl[j] = f2bf(bf[j] - bf2f(bh[j]));
        }
        *(ushort4*)&Ahi[r * SA + c] = make_ushort4(ah[0], ah[1], ah[2], ah[3]);
        *(ushort4*)&Alo[r * SA + c] = make_ushort4(al[0], al[1], al[2], al[3]);
        *(ushort4*)&Bhi[r * SA + c] = make_ushort4(bh[0], bh[1], bh[2], bh[3]);
        *(ushort4*)&Blo[r * SA + c] = make_ushort4(bl[0], bl[1], bl[2], bl[3]);
    }
    __syncthreads();

    // ---- per-wave WMMA: wave w owns tile-row (w>>1), tile-cols {2c, 2c+1} ----
    const int wave = tid >> 5;
    const int lane = tid & 31;
    const int half = lane >> 4;
    const int ln   = lane & 15;
    const int tm   = wave >> 1;
    const int tc0  = (wave & 1) * 2;

    const int mrow = tm * 16 + ln;        // A panel row for this lane
    const int n0   = tc0 * 16 + ln;       // B panel row (= output column) tile 0
    const int n1   = n0 + 16;             // tile 1

    v8f c0 = {0.f, 0.f, 0.f, 0.f, 0.f, 0.f, 0.f, 0.f};
    v8f c1 = {0.f, 0.f, 0.f, 0.f, 0.f, 0.f, 0.f, 0.f};

    #pragma unroll
    for (int kk = 0; kk < 4; ++kk) {
        // A frag (16x32 bf16): lane half h holds K in {8h..8h+7, 16+8h..16+8h+7}
        const int kbA = kk * 32 + half * 8;
        FragBF ah, al;
        ah.u[0] = *(const uint4*)&Ahi[mrow * SA + kbA];
        ah.u[1] = *(const uint4*)&Ahi[mrow * SA + kbA + 16];
        al.u[0] = *(const uint4*)&Alo[mrow * SA + kbA];
        al.u[1] = *(const uint4*)&Alo[mrow * SA + kbA + 16];

        // B frag (32x16 bf16): lane half h holds K in {8h..8h+7, 16+8h..16+8h+7}
        const int kbB = kk * 32 + half * 8;
        FragBF b0h, b0l, b1h, b1l;
        b0h.u[0] = *(const uint4*)&Bhi[n0 * SA + kbB];
        b0h.u[1] = *(const uint4*)&Bhi[n0 * SA + kbB + 16];
        b0l.u[0] = *(const uint4*)&Blo[n0 * SA + kbB];
        b0l.u[1] = *(const uint4*)&Blo[n0 * SA + kbB + 16];
        b1h.u[0] = *(const uint4*)&Bhi[n1 * SA + kbB];
        b1h.u[1] = *(const uint4*)&Bhi[n1 * SA + kbB + 16];
        b1l.u[0] = *(const uint4*)&Blo[n1 * SA + kbB];
        b1l.u[1] = *(const uint4*)&Blo[n1 * SA + kbB + 16];

        // split-bf16 3-pass: hi*hi + hi*lo + lo*hi  (lo*lo negligible)
        c0 = __builtin_amdgcn_wmma_f32_16x16x32_bf16(false, ah.v, false, b0h.v,
                                                     (short)0, c0, false, false);
        c0 = __builtin_amdgcn_wmma_f32_16x16x32_bf16(false, ah.v, false, b0l.v,
                                                     (short)0, c0, false, false);
        c0 = __builtin_amdgcn_wmma_f32_16x16x32_bf16(false, al.v, false, b0h.v,
                                                     (short)0, c0, false, false);
        c1 = __builtin_amdgcn_wmma_f32_16x16x32_bf16(false, ah.v, false, b1h.v,
                                                     (short)0, c1, false, false);
        c1 = __builtin_amdgcn_wmma_f32_16x16x32_bf16(false, ah.v, false, b1l.v,
                                                     (short)0, c1, false, false);
        c1 = __builtin_amdgcn_wmma_f32_16x16x32_bf16(false, al.v, false, b1h.v,
                                                     (short)0, c1, false, false);
    }

    // ---- fused epilogue: d = sqrt(max(xx+yy-2dot,0)); diag weighted -1 ----
    // C/D layout: VGPR r, lane l -> M = r + 8*(l>>4), N = l&15
    float lsum = 0.f;
    const int gn0 = bn * BT + tc0 * 16 + ln;
    const int gn1 = gn0 + 16;
    const float y0 = yy[gn0];
    const float y1 = yy[gn1];
    #pragma unroll
    for (int r = 0; r < 8; ++r) {
        const int gm = bm * BT + tm * 16 + 8 * half + r;
        const float xm = xx[gm];
        float d0 = sqrtf(fmaxf(xm + y0 - 2.f * c0[r], 0.f));
        float d1 = sqrtf(fmaxf(xm + y1 - 2.f * c1[r], 0.f));
        lsum += (gm == gn0) ? -d0 : d0;
        lsum += (gm == gn1) ? -d1 : d1;
    }

    // ---- block reduction + one f64 atomic per workgroup ----
    red[tid] = lsum;
    __syncthreads();
    #pragma unroll
    for (int s = 128; s > 0; s >>= 1) {
        if (tid < s) red[tid] += red[tid + s];
        __syncthreads();
    }
    if (tid == 0) atomicAdd(acc, (double)red[0]);
}

// ---------------------------------------------------------------------------
// Kernel 3: scale and emit scalar loss
// ---------------------------------------------------------------------------
__global__ void distloss_finalize(const double* __restrict__ acc,
                                  float* __restrict__ out)
{
    if (threadIdx.x == 0 && blockIdx.x == 0)
        out[0] = (float)(*acc * (WEIGHT / (double)BDIM));
}

// ---------------------------------------------------------------------------
extern "C" void kernel_launch(void* const* d_in, const int* in_sizes, int n_in,
                              void* d_out, int out_size, void* d_ws, size_t ws_size,
                              hipStream_t stream)
{
    (void)in_sizes; (void)n_in; (void)out_size; (void)ws_size;
    const float* outp = (const float*)d_in[0];
    const float* tgtp = (const float*)d_in[1];

    // workspace: xx[8192] f32 | yy[8192] f32 | f64 accumulator (8B aligned)
    float*  xx  = (float*)d_ws;
    float*  yy  = xx + BDIM;
    double* acc = (double*)((char*)d_ws + 2 * BDIM * sizeof(float));

    distloss_norms<<<2048, 256, 0, stream>>>(outp, tgtp, xx, yy, acc);
    distloss_tile<<<dim3(BDIM / BT, BDIM / BT), 256, 0, stream>>>(
        outp, tgtp, xx, yy, acc);
    distloss_finalize<<<1, 64, 0, stream>>>(acc, (float*)d_out);
}
